// GINModel_70334384439967
// MI455X (gfx1250) — matrix operand order repaired and checked by reading, not compiled, not run on hardware
//
#include <hip/hip_runtime.h>
#include <hip/hip_bf16.h>
#include <stdint.h>

#define N_NODES 50000
#define N_EDGES 600000
#define NODE_DIM 128
#define HIDDEN 512
#define OUT_DIM 128

typedef __attribute__((ext_vector_type(16))) __bf16 v16bf;
typedef __attribute__((ext_vector_type(8)))  float v8f;

// ---------- helpers ----------
__device__ __forceinline__ uint32_t pack2bf16(float a, float b) {
  union { float f; uint32_t u; } x, y;
  x.f = a; y.f = b;
  uint32_t ha = (x.u + 0x7FFFu + ((x.u >> 16) & 1u)) >> 16;  // RNE f32->bf16
  uint32_t hb = (y.u + 0x7FFFu + ((y.u >> 16) & 1u)) >> 16;
  return (ha & 0xFFFFu) | (hb << 16);
}

// ---------- elementwise copy (float4) ----------
__global__ void copy_f32_kernel(const float* __restrict__ src,
                                float* __restrict__ dst, int n4) {
  int i = blockIdx.x * blockDim.x + threadIdx.x;
  if (i < n4) ((float4*)dst)[i] = ((const float4*)src)[i];
}

// ---------- pack contiguous f32 pairs -> bf16x2 (A matrices, pairs along K) ----------
__global__ void pack_pairs_kernel(const float* __restrict__ in,
                                  uint32_t* __restrict__ out, int nPairs) {
  int i = blockIdx.x * blockDim.x + threadIdx.x;
  if (i >= nPairs) return;
  float2 v = ((const float2*)in)[i];
  out[i] = pack2bf16(v.x, v.y);
}

// ---------- pack weights: out[kk*N+n] = bf16x2(W[2kk][n], W[2kk+1][n]) ----------
__global__ void pack_w_kernel(const float* __restrict__ W,
                              uint32_t* __restrict__ out, int KP, int N) {
  int i = blockIdx.x * blockDim.x + threadIdx.x;
  if (i >= KP * N) return;
  int kk = i / N, n = i - kk * N;
  out[i] = pack2bf16(W[(2 * kk) * N + n], W[(2 * kk + 1) * N + n]);
}

// ---------- edge scatter-add: agg[dst] += feat[src] (f32 HW atomics) ----------
__global__ void scatter_add_kernel(const float* __restrict__ feat,
                                   const int* __restrict__ src,
                                   const int* __restrict__ dst,
                                   float* __restrict__ agg,
                                   int nEdges, int dim, int chunkShift) {
  long long t = (long long)blockIdx.x * blockDim.x + threadIdx.x;
  int e = (int)(t >> chunkShift);
  if (e >= nEdges) return;
  int c = ((int)t & ((1 << chunkShift) - 1)) * 4;
  int s = src[e], d = dst[e];
  const float4 v = *(const float4*)(feat + (long long)s * dim + c);
  float* out = agg + (long long)d * dim + c;
  unsafeAtomicAdd(out + 0, v.x);   // global_atomic_add_f32
  unsafeAtomicAdd(out + 1, v.y);
  unsafeAtomicAdd(out + 2, v.z);
  unsafeAtomicAdd(out + 3, v.w);
}

// ---------- WMMA bf16 GEMM with async global->LDS staging ----------
// C[M,N] = act(A[M,K] @ W[K,N] + bias), A/W pre-packed as bf16 pairs along K.
// Block = 128 threads (4 waves). Block tile 16(M) x 128(N); each wave owns two
// 16x16 tiles sharing one A fragment -> 2x v_wmma_f32_16x16x32_bf16 per K-step.
// Tiles staged to LDS with global_load_async_to_lds_{b64,b128} (ASYNCcnt),
// double-buffered: issue tile k+1, s_wait_asynccnt 5 for tile k, barrier, MMA.
template <int K, int N, bool RELU>
__global__ __launch_bounds__(128) void gemm_async_wmma_kernel(
    const uint32_t* __restrict__ Abf,   // [M][K/2]  bf16 pairs
    const uint32_t* __restrict__ Wbf,   // [K/2][N]  bf16 pairs
    const float* __restrict__ bias, float* __restrict__ C) {
  constexpr int KP = K / 2;
  constexpr int STEPS = K / 32;
  __shared__ __align__(16) uint32_t ldsA[2][16][16];    // 16 rows x 16 k-pairs
  __shared__ __align__(16) uint32_t ldsB[2][16][128];   // 16 k-pairs x 128 cols

  const int tid  = threadIdx.x;
  const int wave = tid >> 5;
  const int lane = tid & 31;
  const int lh   = lane & 15;          // lane low-half index
  const int hg   = lane >> 4;          // lane group 0/1
  const int m0   = blockIdx.y * 16;
  const int n0   = blockIdx.x * 128;
  const int wn0  = wave * 32;          // this wave's 32-col slice

  // async A: thread -> (row = tid>>3, 8-byte chunk = tid&7); b64 per thread
  const int ar = tid >> 3, ac = tid & 7;
  const uint64_t aBase = (uint64_t)(uintptr_t)Abf;
  const uint32_t aOff0 = (uint32_t)((((m0 + ar) * KP) + ac * 2) * 4);
  // async B: 4x b128 per thread over 16 rows x 512B
  const uint64_t bBase = (uint64_t)(uintptr_t)Wbf;
  uint32_t bOff0[4];
  #pragma unroll
  for (int j = 0; j < 4; ++j) {
    int l16 = tid + j * 128;                       // 0..511 16B-chunks
    int br = l16 >> 5, bc = l16 & 31;
    bOff0[j] = (uint32_t)(((br * N) + n0 + bc * 4) * 4);
  }

  auto issueTile = [&](int buf, int ks) {
    uint32_t av = aOff0 + (uint32_t)(ks * 64);                  // 16 pairs * 4B
    uint32_t al = (uint32_t)(uintptr_t)&ldsA[buf][ar][ac * 2];
    asm volatile("global_load_async_to_lds_b64 %0, %1, %2"
                 :: "v"(al), "v"(av), "s"(aBase) : "memory");
    #pragma unroll
    for (int j = 0; j < 4; ++j) {
      int l16 = tid + j * 128;
      int br = l16 >> 5, bc = l16 & 31;
      uint32_t bv = bOff0[j] + (uint32_t)(ks * (16 * N * 4));
      uint32_t bl = (uint32_t)(uintptr_t)&ldsB[buf][br][bc * 4];
      asm volatile("global_load_async_to_lds_b128 %0, %1, %2"
                   :: "v"(bl), "v"(bv), "s"(bBase) : "memory");
    }
  };

  v8f acc0 = {0.f, 0.f, 0.f, 0.f, 0.f, 0.f, 0.f, 0.f};
  v8f acc1 = {0.f, 0.f, 0.f, 0.f, 0.f, 0.f, 0.f, 0.f};

  issueTile(0, 0);
  for (int s = 0; s < STEPS; ++s) {
    const int buf = s & 1;
    if (s + 1 < STEPS) {
      issueTile(buf ^ 1, s + 1);
      asm volatile("s_wait_asynccnt %0" :: "i"(5) : "memory");  // tile s done
    } else {
      asm volatile("s_wait_asynccnt %0" :: "i"(0) : "memory");
    }
    __syncthreads();

    union { v16bf v; uint32_t u[8]; } aF, bF0, bF1;
    // A frag: row=lh, lanes 0-15 K=0..15, lanes 16-31 K=16..31 (matches B)
    #pragma unroll
    for (int i = 0; i < 4; ++i) aF.u[i] = ldsA[buf][lh][hg * 8 + i];
    #pragma unroll
    for (int i = 4; i < 8; ++i) aF.u[i] = ldsA[buf][lh][hg * 8 + i];
    // B frags: lanes 0-15 K=0..15, lanes 16-31 K=16..31; N = lh
    #pragma unroll
    for (int i = 0; i < 8; ++i) bF0.u[i] = ldsB[buf][hg * 8 + i][wn0 + lh];
    #pragma unroll
    for (int i = 0; i < 8; ++i) bF1.u[i] = ldsB[buf][hg * 8 + i][wn0 + 16 + lh];

    acc0 = __builtin_amdgcn_wmma_f32_16x16x32_bf16(false, aF.v, false, bF0.v,
                                                   (short)0, acc0, false, false);
    acc1 = __builtin_amdgcn_wmma_f32_16x16x32_bf16(false, aF.v, false, bF1.v,
                                                   (short)0, acc1, false, false);
    __syncthreads();
  }

  // C/D layout: VGPR r -> row = m0 + hg*8 + r; col = lh (M multiple of 16, no guards)
  const int c0 = n0 + wn0 + lh;
  const float b0 = bias[c0], b1 = bias[c0 + 16];
  #pragma unroll
  for (int r = 0; r < 8; ++r) {
    const int row = m0 + hg * 8 + r;
    float v0 = acc0[r] + b0;
    float v1 = acc1[r] + b1;
    if constexpr (RELU) { v0 = fmaxf(v0, 0.f); v1 = fmaxf(v1, 0.f); }
    C[row * N + c0]      = v0;
    C[row * N + c0 + 16] = v1;
  }
}

// ---------- column sum for mean pooling ----------
__global__ __launch_bounds__(512) void colsum_kernel(const float* __restrict__ h,
                                                     float* __restrict__ pooled,
                                                     int nRows, int rowsPerBlock) {
  int c  = threadIdx.x;                 // 0..511
  int r0 = blockIdx.x * rowsPerBlock;
  int r1 = min(r0 + rowsPerBlock, nRows);
  float s = 0.f;
  for (int r = r0; r < r1; ++r) s += h[(long long)r * HIDDEN + c];
  unsafeAtomicAdd(&pooled[c], s);
}

__global__ void zero_f32_kernel(float* __restrict__ p, int n) {
  int i = blockIdx.x * blockDim.x + threadIdx.x;
  if (i < n) p[i] = 0.f;
}

// ---------- final FC: out = (pooled/N) @ Wf + bf ----------
__global__ __launch_bounds__(128) void final_fc_kernel(
    const float* __restrict__ pooled, const float* __restrict__ Wf,
    const float* __restrict__ bf, float* __restrict__ out) {
  __shared__ float p[HIDDEN];
  for (int i = threadIdx.x; i < HIDDEN; i += 128) p[i] = pooled[i] * (1.0f / N_NODES);
  __syncthreads();
  int j = threadIdx.x;                  // 0..127
  float s = bf[j];
  for (int k = 0; k < HIDDEN; ++k) s = fmaf(p[k], Wf[k * OUT_DIM + j], s);
  out[j] = s;
}

// ---------- launcher ----------
extern "C" void kernel_launch(void* const* d_in, const int* in_sizes, int n_in,
                              void* d_out, int out_size, void* d_ws, size_t ws_size,
                              hipStream_t stream) {
  const float* x   = (const float*)d_in[0];
  const int*   ei  = (const int*)d_in[1];
  const int*   src = ei;               // edge_index[0]
  const int*   dst = ei + N_EDGES;     // edge_index[1]
  const float* W1a = (const float*)d_in[2];
  const float* b1a = (const float*)d_in[3];
  const float* W1b = (const float*)d_in[4];
  const float* b1b = (const float*)d_in[5];
  const float* W2a = (const float*)d_in[6];
  const float* b2a = (const float*)d_in[7];
  const float* W2b = (const float*)d_in[8];
  const float* b2b = (const float*)d_in[9];
  const float* Wf  = (const float*)d_in[10];
  const float* bfv = (const float*)d_in[11];
  float* out = (float*)d_out;

  float* ws     = (float*)d_ws;
  float* agg1   = ws;                                      // [N, 128] f32
  float* t      = agg1 + (size_t)N_NODES * NODE_DIM;       // [N, 512] f32
  float* h1     = t    + (size_t)N_NODES * HIDDEN;         // [N, 512] f32
  float* agg2   = h1   + (size_t)N_NODES * HIDDEN;         // [N, 512] f32
  float* pooled = agg2 + (size_t)N_NODES * HIDDEN;         // [512]
  uint32_t* abf   = (uint32_t*)(pooled + HIDDEN);          // [N, 512/2] pairs (max)
  uint32_t* w1aB  = abf  + (size_t)N_NODES * (HIDDEN / 2);
  uint32_t* w1bB  = w1aB + (NODE_DIM / 2) * HIDDEN;
  uint32_t* w2aB  = w1bB + (HIDDEN / 2) * HIDDEN;
  uint32_t* w2bB  = w2aB + (HIDDEN / 2) * HIDDEN;
  float* h2 = h1;  // h1 dead after conv2 aggregation + first conv2 GEMM

  const dim3 blk256(256);

  // ---- one-time weight packing to bf16 pair layout ----
  {
    int n1 = (NODE_DIM / 2) * HIDDEN;
    pack_w_kernel<<<(n1 + 255) / 256, blk256, 0, stream>>>(W1a, w1aB, NODE_DIM / 2, HIDDEN);
    int n2 = (HIDDEN / 2) * HIDDEN;
    pack_w_kernel<<<(n2 + 255) / 256, blk256, 0, stream>>>(W1b, w1bB, HIDDEN / 2, HIDDEN);
    pack_w_kernel<<<(n2 + 255) / 256, blk256, 0, stream>>>(W2a, w2aB, HIDDEN / 2, HIDDEN);
    pack_w_kernel<<<(n2 + 255) / 256, blk256, 0, stream>>>(W2b, w2bB, HIDDEN / 2, HIDDEN);
  }

  // ---- conv1: agg1 = x + segment_sum(x[src] -> dst) ----
  {
    int n4 = N_NODES * NODE_DIM / 4;
    copy_f32_kernel<<<(n4 + 255) / 256, blk256, 0, stream>>>(x, agg1, n4);
    long long threads = (long long)N_EDGES * (NODE_DIM / 4);
    scatter_add_kernel<<<(unsigned)((threads + 255) / 256), blk256, 0, stream>>>(
        x, src, dst, agg1, N_EDGES, NODE_DIM, 5);
  }
  // t = relu(agg1 @ W1a + b1a);  h1 = t @ W1b + b1b
  {
    int np1 = N_NODES * NODE_DIM / 2;
    pack_pairs_kernel<<<(np1 + 255) / 256, blk256, 0, stream>>>(agg1, abf, np1);
    dim3 grid(HIDDEN / 128, N_NODES / 16);
    gemm_async_wmma_kernel<NODE_DIM, HIDDEN, true>
        <<<grid, 128, 0, stream>>>(abf, w1aB, b1a, t);
    int np2 = N_NODES * HIDDEN / 2;
    pack_pairs_kernel<<<(np2 + 255) / 256, blk256, 0, stream>>>(t, abf, np2);
    gemm_async_wmma_kernel<HIDDEN, HIDDEN, false>
        <<<grid, 128, 0, stream>>>(abf, w1bB, b1b, h1);
  }

  // ---- conv2: agg2 = h1 + segment_sum(h1[src] -> dst) ----
  {
    int n4 = N_NODES * HIDDEN / 4;
    copy_f32_kernel<<<(n4 + 255) / 256, blk256, 0, stream>>>(h1, agg2, n4);
    long long threads = (long long)N_EDGES * (HIDDEN / 4);
    scatter_add_kernel<<<(unsigned)((threads + 255) / 256), blk256, 0, stream>>>(
        h1, src, dst, agg2, N_EDGES, HIDDEN, 7);
  }
  // t = relu(agg2 @ W2a + b2a);  h2 = t @ W2b + b2b
  {
    int np2 = N_NODES * HIDDEN / 2;
    pack_pairs_kernel<<<(np2 + 255) / 256, blk256, 0, stream>>>(agg2, abf, np2);
    dim3 grid(HIDDEN / 128, N_NODES / 16);
    gemm_async_wmma_kernel<HIDDEN, HIDDEN, true>
        <<<grid, 128, 0, stream>>>(abf, w2aB, b2a, t);
    pack_pairs_kernel<<<(np2 + 255) / 256, blk256, 0, stream>>>(t, abf, np2);
    gemm_async_wmma_kernel<HIDDEN, HIDDEN, false>
        <<<grid, 128, 0, stream>>>(abf, w2bB, b2b, h2);
  }

  // ---- mean pool + final FC ----
  zero_f32_kernel<<<1, 512, 0, stream>>>(pooled, HIDDEN);
  {
    int rowsPerBlock = 256;
    int nBlocks = (N_NODES + rowsPerBlock - 1) / rowsPerBlock;
    colsum_kernel<<<nBlocks, 512, 0, stream>>>(h2, pooled, N_NODES, rowsPerBlock);
  }
  final_fc_kernel<<<1, 128, 0, stream>>>(pooled, Wf, bfv, out);
}
